// Attention_56444460204293
// MI455X (gfx1250) — compile-verified, matrix operand order checked by reading, not run on hardware
//
#include <hip/hip_runtime.h>
#include <hip/hip_bf16.h>

#define DIM     1024
#define NHEADS  16
#define HDIM    64
#define SEQ     2048
#define BATCH   4
#define NTOK    (BATCH * SEQ)      /* 8192 */
#define QK_SCALE 0.125f            /* 64^-0.5 */

typedef __attribute__((ext_vector_type(16))) __bf16        v16bf;
typedef __attribute__((ext_vector_type(8)))  float         v8f;
typedef __attribute__((ext_vector_type(8)))  unsigned short v8u16;

union Frag16 { v8u16 u[2]; v16bf bf; };

__device__ __forceinline__ unsigned short f2bf(float f) {
  unsigned int u = __builtin_bit_cast(unsigned int, f);
  u += 0x7FFFu + ((u >> 16) & 1u);          // round-to-nearest-even
  return (unsigned short)(u >> 16);
}

__device__ __forceinline__ v8f vzero8() {
  v8f z = {0.f, 0.f, 0.f, 0.f, 0.f, 0.f, 0.f, 0.f};
  return z;
}

// Wave-relative LDS byte address (flat aperture keeps the offset in addr[31:0]).
__device__ __forceinline__ unsigned lds_u32(const void* p) {
  return (unsigned)(size_t)p;
}

// CDNA5 async memory->LDS copy, 16B per lane.  Tracked by ASYNCcnt.
__device__ __forceinline__ void async_ld16(unsigned lds_addr,
                                           const unsigned short* g) {
  asm volatile("global_load_async_to_lds_b128 %0, %1, off"
               :: "v"(lds_addr), "v"(g)
               : "memory");
}
__device__ __forceinline__ void wait_async0() {
  asm volatile("s_wait_asynccnt 0x0" ::: "memory");
}
__device__ __forceinline__ void wait_async4() {
  asm volatile("s_wait_asynccnt 0x4" ::: "memory");
}

// Load a 16x32 bf16 fragment (A layout; B layout is symmetric with lane = N
// column when the source tile is stored [n][k] row-major).
// Lane l (l<16): row l, K = 0..7 and 16..23 ; lane l>=16: row l-16, K = 8..15, 24..31.
__device__ __forceinline__ v16bf frag_ld(const unsigned short* base, int ld, int lane) {
  const int r  = lane & 15;
  const int kb = (lane & 16) >> 1;          // 0 or 8
  const unsigned short* p = base + r * ld + kb;
  Frag16 f;
  f.u[0] = *(const v8u16*)(p);              // K = kb .. kb+7
  f.u[1] = *(const v8u16*)(p + 16);         // K = kb+16 .. kb+23
  return f.bf;
}

__device__ __forceinline__ v8f wmma_bf16(v16bf a, v16bf b, v8f c) {
  return __builtin_amdgcn_wmma_f32_16x16x32_bf16(false, a, false, b, (short)0, c, false, false);
}

// ---------------------------------------------------------------------------
// f32 -> bf16 conversion
// ---------------------------------------------------------------------------
__global__ void k_f32_to_bf16(const float* __restrict__ src,
                              unsigned short* __restrict__ dst, int n) {
  int i = (blockIdx.x * blockDim.x + threadIdx.x) * 4;
  if (i + 3 < n) {
    dst[i + 0] = f2bf(src[i + 0]);
    dst[i + 1] = f2bf(src[i + 1]);
    dst[i + 2] = f2bf(src[i + 2]);
    dst[i + 3] = f2bf(src[i + 3]);
  }
}

// ---------------------------------------------------------------------------
// Tiled bf16 WMMA GEMM:  C[m,n] = sum_k A[m,k] * Bw[n,k]
// Block tile 128x128, BK=32, 256 threads = 8 waves; wave -> 32x64 subtile.
// Double-buffered LDS fed by async global->LDS copies (ASYNCcnt).
// MODE 0: QKV epilogue: q scaled, q/k row-major [bh][SEQ][HDIM],
//                       v stored TRANSPOSED [bh][HDIM][SEQ] (B-operand layout).
// MODE 1: proj epilogue (f32 + bias -> d_out)
// ---------------------------------------------------------------------------
template <int MODE>
__global__ __launch_bounds__(256) void k_gemm(
    const unsigned short* __restrict__ A,     // [M,K] bf16
    const unsigned short* __restrict__ Bw,    // [N,K] bf16
    unsigned short* __restrict__ qkv_out,     // MODE 0
    float* __restrict__ f_out,                // MODE 1
    const float* __restrict__ bias,           // MODE 1
    int M, int N, int K) {
  constexpr int BK = 32, LDT = 40;            // 40 shorts = 80B (16B multiple)
  __shared__ __attribute__((aligned(16))) unsigned short lA[2][128 * LDT];
  __shared__ __attribute__((aligned(16))) unsigned short lB[2][128 * LDT];

  const int t    = threadIdx.x;
  const int lane = t & 31;
  const int wid  = t >> 5;
  const int wm   = wid & 3;                   // 4 waves along M
  const int wn   = wid >> 2;                  // 2 waves along N
  const int m0   = blockIdx.x * 128;
  const int n0   = blockIdx.y * 128;

  v8f acc[2][4];
#pragma unroll
  for (int i = 0; i < 2; ++i)
#pragma unroll
    for (int j = 0; j < 4; ++j) acc[i][j] = vzero8();

  // cooperative tile load mapping: 2 threads per 32-element row slice
  const int lrow = t >> 1;
  const int lcol = (t & 1) * 16;
  const unsigned short* gA = A  + (size_t)(m0 + lrow) * K + lcol;
  const unsigned short* gB = Bw + (size_t)(n0 + lrow) * K + lcol;
  unsigned sA[2], sB[2];
  sA[0] = lds_u32(&lA[0][lrow * LDT + lcol]);
  sA[1] = lds_u32(&lA[1][lrow * LDT + lcol]);
  sB[0] = lds_u32(&lB[0][lrow * LDT + lcol]);
  sB[1] = lds_u32(&lB[1][lrow * LDT + lcol]);

  auto issue = [&](int buf, int koff) {
    async_ld16(sA[buf],      gA + koff);
    async_ld16(sA[buf] + 16, gA + koff + 8);
    async_ld16(sB[buf],      gB + koff);
    async_ld16(sB[buf] + 16, gB + koff + 8);
  };

  issue(0, 0);                                // prologue prefetch
  for (int k0 = 0; k0 < K; k0 += BK) {
    const int buf  = (k0 >> 5) & 1;
    const bool more = (k0 + BK) < K;
    if (more) issue(buf ^ 1, k0 + BK);        // prefetch next tile
    if (more) wait_async4(); else wait_async0();
    __syncthreads();

    v16bf af[2], bfr[4];
#pragma unroll
    for (int i = 0; i < 2; ++i)
      af[i] = frag_ld(&lA[buf][(32 * wm + 16 * i) * LDT], LDT, lane);
#pragma unroll
    for (int j = 0; j < 4; ++j)
      bfr[j] = frag_ld(&lB[buf][(64 * wn + 16 * j) * LDT], LDT, lane);
#pragma unroll
    for (int i = 0; i < 2; ++i)
#pragma unroll
      for (int j = 0; j < 4; ++j)
        acc[i][j] = wmma_bf16(af[i], bfr[j], acc[i][j]);
    __syncthreads();
  }

  // Epilogue.  C layout: col = lane&15, row = vgpr + 8*(lane>=16).
  const int rbase = (lane >> 4) << 3;
  const int cl    = lane & 15;
#pragma unroll
  for (int i = 0; i < 2; ++i) {
#pragma unroll
    for (int j = 0; j < 4; ++j) {
      const int gm0 = m0 + 32 * wm + 16 * i + rbase;
      const int gn  = n0 + 64 * wn + 16 * j + cl;
      if (MODE == 0) {
        const int kind = gn >> 10;            // 0=q 1=k 2=v
        const int h    = (gn >> 6) & (NHEADS - 1);
        const int d    = gn & (HDIM - 1);
        const float sc = (kind == 0) ? QK_SCALE : 1.0f;
#pragma unroll
        for (int r = 0; r < 8; ++r) {
          const int gm = gm0 + r;
          const int b  = gm >> 11;            // / SEQ
          const int n  = gm & (SEQ - 1);
          const size_t cbase =
              ((size_t)(b * NHEADS + h) * 3 + kind) * ((size_t)SEQ * HDIM);
          const size_t off = (kind == 2)
              ? (cbase + (size_t)d * SEQ + n)      // V^T: [d][n]
              : (cbase + (size_t)n * HDIM + d);    // Q/K: [n][d]
          qkv_out[off] = f2bf(acc[i][j][r] * sc);
        }
      } else {
        const float bb = bias[gn];
#pragma unroll
        for (int r = 0; r < 8; ++r)
          f_out[(size_t)(gm0 + r) * N + gn] = acc[i][j][r] + bb;
      }
    }
  }
}

// ---------------------------------------------------------------------------
// Flash attention: block = 128 queries of one (b,h); wave = 16 query rows.
// Streams K (row-major) and V^T chunks of 64 keys with async double buffering
// and online softmax.
// ---------------------------------------------------------------------------
__global__ __launch_bounds__(256) void k_attn(
    const unsigned short* __restrict__ qkv,   // [bh][3] : q/k [SEQ][HDIM], v [HDIM][SEQ]
    unsigned short* __restrict__ outbf) {     // [B][SEQ][DIM] bf16
  constexpr int LD    = 72;                   // 72 shorts = 144B (16B multiple)
  constexpr int CHUNK = 64 * LD;              // one 64-row chunk (shorts)
  // Pool slots: [0..1] = K double buffer, [2..3] = V^T double buffer.
  // Q staging aliases the front (slots 0..1) before the stream loop starts.
  __shared__ __attribute__((aligned(16))) unsigned short pool[4 * CHUNK];
  __shared__ __attribute__((aligned(16))) unsigned short lP[8 * 16 * LD];

  const int t    = threadIdx.x;
  const int lane = t & 31;
  const int wid  = t >> 5;
  const int bh   = blockIdx.y;                // 0..63
  const int q0   = blockIdx.x * 128;

  const size_t base = (size_t)bh * 3 * SEQ * HDIM;
  const unsigned short* Qp = qkv + base;                            // [SEQ][HDIM]
  const unsigned short* Kp = qkv + base + (size_t)SEQ * HDIM;       // [SEQ][HDIM]
  const unsigned short* Vp = qkv + base + (size_t)2 * SEQ * HDIM;   // [HDIM][SEQ]

  // stage Q tile (128 x 64) via async copies into the pool front
  {
    const int row = t >> 1, ch = (t & 1) * 32;
    const unsigned short* g = Qp + (size_t)(q0 + row) * HDIM + ch;
    const unsigned sq = lds_u32(&pool[row * LD + ch]);
#pragma unroll
    for (int u = 0; u < 4; ++u)
      async_ld16(sq + 16 * u, g + 8 * u);
  }
  wait_async0();
  __syncthreads();
  v16bf qf[2];
  qf[0] = frag_ld(&pool[(16 * wid) * LD], LD, lane);
  qf[1] = frag_ld(&pool[(16 * wid) * LD + 32], LD, lane);
  __syncthreads();                            // all waves done reading Q (pool reuse)

  float mrow[8], lrow[8];
#pragma unroll
  for (int r = 0; r < 8; ++r) { mrow[r] = -1e30f; lrow[r] = 0.f; }
  v8f oacc[4];
#pragma unroll
  for (int j = 0; j < 4; ++j) oacc[j] = vzero8();

  const int krow = t >> 2, kch = (t & 3) * 16;
  const int cl = lane & 15;
  unsigned short* Pw = &lP[wid * 16 * LD];
  unsigned sK[2], sV[2];
  sK[0] = lds_u32(&pool[0 * CHUNK + krow * LD + kch]);
  sK[1] = lds_u32(&pool[1 * CHUNK + krow * LD + kch]);
  sV[0] = lds_u32(&pool[2 * CHUNK + krow * LD + kch]);
  sV[1] = lds_u32(&pool[3 * CHUNK + krow * LD + kch]);

  auto issueKV = [&](int buf, int kv) {
    const unsigned short* gk = Kp + (size_t)(kv + krow) * HDIM + kch; // K rows
    async_ld16(sK[buf],      gk);
    async_ld16(sK[buf] + 16, gk + 8);
    const unsigned short* gv = Vp + (size_t)krow * SEQ + kv + kch;    // V^T rows
    async_ld16(sV[buf],      gv);
    async_ld16(sV[buf] + 16, gv + 8);
  };

  issueKV(0, 0);                              // prologue prefetch
  int it = 0;
  for (int kv0 = 0; kv0 < SEQ; kv0 += 64, ++it) {
    const int buf   = it & 1;
    const bool more = (kv0 + 64) < SEQ;
    if (more) issueKV(buf ^ 1, kv0 + 64);     // prefetch next chunk
    if (more) wait_async4(); else wait_async0();
    __syncthreads();

    const unsigned short* lK  = &pool[buf * CHUNK];
    const unsigned short* lVt = &pool[(2 + buf) * CHUNK];

    // S = Q * K^T  (16 x 64 per wave)
    v8f sacc[4];
#pragma unroll
    for (int j = 0; j < 4; ++j) sacc[j] = vzero8();
#pragma unroll
    for (int ks = 0; ks < 2; ++ks)
#pragma unroll
      for (int j = 0; j < 4; ++j)
        sacc[j] = wmma_bf16(qf[ks],
                            frag_ld(lK + (16 * j) * LD + 32 * ks, LD, lane),
                            sacc[j]);

    // online softmax (row = vgpr r + 8*(lane>=16); 16 lanes share a row)
#pragma unroll
    for (int r = 0; r < 8; ++r) {
      float smax = sacc[0][r];
#pragma unroll
      for (int j = 1; j < 4; ++j) smax = fmaxf(smax, sacc[j][r]);
#pragma unroll
      for (int m = 8; m >= 1; m >>= 1)
        smax = fmaxf(smax, __shfl_xor(smax, m, 32));
      const float mnew  = fmaxf(mrow[r], smax);
      const float alpha = __expf(mrow[r] - mnew);
      mrow[r] = mnew;
      float psum = 0.f;
      const int rl = r + ((lane >> 4) << 3);
#pragma unroll
      for (int j = 0; j < 4; ++j) {
        const float p = __expf(sacc[j][r] - mnew);
        psum += p;
        Pw[rl * LD + 16 * j + cl] = f2bf(p);   // C-layout -> A-layout via LDS
      }
#pragma unroll
      for (int m = 8; m >= 1; m >>= 1) psum += __shfl_xor(psum, m, 32);
      lrow[r] = lrow[r] * alpha + psum;
#pragma unroll
      for (int j = 0; j < 4; ++j) oacc[j][r] *= alpha;
    }

    // O += P * V   (wave-private P; in-wave LDS ordering suffices)
#pragma unroll
    for (int ks = 0; ks < 2; ++ks) {
      v16bf pf = frag_ld(&Pw[32 * ks], LD, lane);
#pragma unroll
      for (int j = 0; j < 4; ++j)
        oacc[j] = wmma_bf16(pf,
                            frag_ld(lVt + (16 * j) * LD + 32 * ks, LD, lane),
                            oacc[j]);
    }
    __syncthreads();
  }

  // normalize + write [B][SEQ][DIM] bf16
  const int b = bh >> 4, h = bh & (NHEADS - 1);
#pragma unroll
  for (int j = 0; j < 4; ++j)
#pragma unroll
    for (int r = 0; r < 8; ++r) {
      const int n = q0 + 16 * wid + r + ((lane >> 4) << 3);
      const float v = oacc[j][r] / lrow[r];
      outbf[((size_t)(b * SEQ + n)) * DIM + h * HDIM + 16 * j + cl] = f2bf(v);
    }
}

// ---------------------------------------------------------------------------
extern "C" void kernel_launch(void* const* d_in, const int* in_sizes, int n_in,
                              void* d_out, int out_size, void* d_ws, size_t ws_size,
                              hipStream_t stream) {
  (void)in_sizes; (void)n_in; (void)out_size; (void)ws_size;
  const float* x      = (const float*)d_in[0];   // [4,2048,1024]
  const float* w_qkv  = (const float*)d_in[1];   // [3072,1024]
  const float* w_proj = (const float*)d_in[2];   // [1024,1024]
  const float* b_proj = (const float*)d_in[3];   // [1024]
  float* out = (float*)d_out;                    // [4,2048,1024]

  unsigned short* ws      = (unsigned short*)d_ws;
  unsigned short* xbf     = ws;                                   // 8192*1024
  unsigned short* wqkvbf  = xbf    + (size_t)NTOK * DIM;          // 3072*1024
  unsigned short* wprojbf = wqkvbf + (size_t)3 * DIM * DIM;       // 1024*1024
  unsigned short* qkvbf   = wprojbf + (size_t)DIM * DIM;          // 3*8192*1024
  unsigned short* aobf    = qkvbf  + (size_t)3 * NTOK * DIM;      // 8192*1024

  const int n_x = NTOK * DIM, n_wq = 3 * DIM * DIM, n_wp = DIM * DIM;
  k_f32_to_bf16<<<(n_x  / 4 + 255) / 256, 256, 0, stream>>>(x,      xbf,     n_x);
  k_f32_to_bf16<<<(n_wq / 4 + 255) / 256, 256, 0, stream>>>(w_qkv,  wqkvbf,  n_wq);
  k_f32_to_bf16<<<(n_wp / 4 + 255) / 256, 256, 0, stream>>>(w_proj, wprojbf, n_wp);

  // QKV projection: [8192,1024] x [3072,1024]^T
  k_gemm<0><<<dim3(NTOK / 128, (3 * DIM) / 128), 256, 0, stream>>>(
      xbf, wqkvbf, qkvbf, nullptr, nullptr, NTOK, 3 * DIM, DIM);

  // Attention: 16 q-tiles x 64 (b,h) pairs
  k_attn<<<dim3(SEQ / 128, BATCH * NHEADS), 256, 0, stream>>>(qkvbf, aobf);

  // Output projection: [8192,1024] x [1024,1024]^T + bias
  k_gemm<1><<<dim3(NTOK / 128, DIM / 128), 256, 0, stream>>>(
      aobf, wprojbf, nullptr, out, b_proj, NTOK, DIM, DIM);
}
